// AutoCompleteDecoderModel_37185826849356
// MI455X (gfx1250) — matrix operand order repaired and checked by reading, not compiled, not run on hardware
//
#include <hip/hip_runtime.h>
#include <math.h>

#define B_SZ 256
#define S_SZ 512
#define H_SZ 512
#define V_SZ 128
#define G_SZ 2048               // 4*H
#define TILE_M 16
#define N_TILES (B_SZ / TILE_M) // 16 workgroups
#define N_WAVES 16
#define N_THREADS (N_WAVES * 32)

typedef __attribute__((ext_vector_type(16))) __bf16 v16bf;
typedef __attribute__((ext_vector_type(8)))  __bf16 v8bf;
typedef __attribute__((ext_vector_type(8)))  float  v8f;

__device__ __forceinline__ unsigned short f2bf(float f) {
  union { float f; unsigned u; } x; x.f = f;
  unsigned r = (x.u + 0x7FFFu + ((x.u >> 16) & 1u)) >> 16; // RNE
  return (unsigned short)r;
}
__device__ __forceinline__ float sigmf(float x) { return 1.0f / (1.0f + __expf(-x)); }

// 16-bit A-matrix 16x32 fragment from LDS h[16][512] (row-major).
// Lane layout: lanes 0-15 M=lane, K = kb..kb+15 contiguous; kb = (lane>=16)?16:0.
__device__ __forceinline__ v16bf load_a_frag(const unsigned short* hbuf, int lane, int kk) {
  const int m  = lane & 15;
  const int kb = (lane >> 4) << 4;
  const int k0 = kk * 32 + kb;
  v8bf lo = *(const v8bf*)(hbuf + m * H_SZ + k0);
  v8bf hi = *(const v8bf*)(hbuf + m * H_SZ + k0 + 8);
  return __builtin_shufflevector(lo, hi, 0,1,2,3,4,5,6,7,8,9,10,11,12,13,14,15);
}

// One LSTM step for this wave's 32 hidden units (u0..u0+31) of a 16-row batch tile.
// acc init = bias[j] + W_ih[j][x_idx[m]]  (one-hot input == column gather)
// then 16x K-accumulated bf16 WMMAs per gate subtile; pointwise update in registers.
__device__ __forceinline__ void lstm_wave_step(
    const unsigned short* __restrict__ Whh,   // [2048*512] bf16, row-major (gate j, hidden k)
    const float* __restrict__ bias,           // [2048] = b_ih + b_hh
    const float* __restrict__ Wih,            // [2048*128] fp32
    const int* __restrict__ idx_lds,          // [16] current input char per row
    const unsigned short* __restrict__ hbuf,  // [16*512] bf16 h_prev
    int lane, int wave,
    float (&c)[2][8], float (&hout)[2][8])
{
  const int n   = lane & 15;
  const int mhi = (lane >> 4) << 3;           // C/D: m = r + mhi
  const int kb  = (lane >> 4) << 4;           // B: K group base 0/16
  const int u0  = wave * 32;

  v8f acc[4][2];
  #pragma unroll
  for (int q = 0; q < 4; q++)
    #pragma unroll
    for (int t = 0; t < 2; t++) {
      const int j = q * H_SZ + u0 + t * 16 + n;   // gate row (i,f,g,o blocks of H)
      const float base = bias[j];
      #pragma unroll
      for (int r = 0; r < 8; r++)
        acc[q][t][r] = base + Wih[j * V_SZ + idx_lds[r + mhi]];
    }

  #pragma unroll 4
  for (int kk = 0; kk < 16; kk++) {
    v16bf a = load_a_frag(hbuf, lane, kk);
    #pragma unroll
    for (int q = 0; q < 4; q++)
      #pragma unroll
      for (int t = 0; t < 2; t++) {
        const int j = q * H_SZ + u0 + t * 16 + n;
        v16bf b = *(const v16bf*)(Whh + (size_t)j * H_SZ + kk * 32 + kb);
        acc[q][t] = __builtin_amdgcn_wmma_f32_16x16x32_bf16(
            false, a, false, b, (short)0, acc[q][t], false, false);
      }
  }

  #pragma unroll
  for (int t = 0; t < 2; t++)
    #pragma unroll
    for (int r = 0; r < 8; r++) {
      float iv = sigmf(acc[0][t][r]);
      float fv = sigmf(acc[1][t][r]);
      float gv = tanhf(acc[2][t][r]);
      float ov = sigmf(acc[3][t][r]);
      float cv = fv * c[t][r] + iv * gv;
      c[t][r] = cv;
      hout[t][r] = ov * tanhf(cv);
    }
}

__global__ void __launch_bounds__(N_THREADS, 1)
seq2seq_loss_kernel(const int* __restrict__ C_idx, const int* __restrict__ E,
                    const float* __restrict__ encWih, const float* __restrict__ decWih,
                    const unsigned short* __restrict__ encWhh,
                    const unsigned short* __restrict__ decWhh,
                    const unsigned short* __restrict__ projW,
                    const float* __restrict__ biasE, const float* __restrict__ biasD,
                    const float* __restrict__ projB,
                    float* __restrict__ partial)
{
  __shared__ unsigned short hbuf[TILE_M * H_SZ];  // h state, bf16
  __shared__ int   idx_lds[TILE_M];
  __shared__ float logit_lds[TILE_M * V_SZ];
  __shared__ float rownll[TILE_M];
  __shared__ float rowcnt[TILE_M];

  const int tid  = threadIdx.x;
  const int lane = tid & 31;
  const int wave = tid >> 5;
  const int tile = blockIdx.x;
  const int b0   = tile * TILE_M;

  for (int i = tid; i < TILE_M * H_SZ; i += N_THREADS) hbuf[i] = 0;

  float c[2][8], hout[2][8];
  #pragma unroll
  for (int t = 0; t < 2; t++)
    #pragma unroll
    for (int r = 0; r < 8; r++) c[t][r] = 0.0f;

  const int n   = lane & 15;
  const int mhi = (lane >> 4) << 3;
  const int kb  = (lane >> 4) << 4;
  const int u0  = wave * 32;

  // ---------------- Encoder: 512 recurrent steps ----------------
  for (int s = 0; s < S_SZ; s++) {
    if (tid < TILE_M) idx_lds[tid] = C_idx[(b0 + tid) * S_SZ + s];
    __syncthreads();                                  // h_prev + idx visible
    lstm_wave_step(encWhh, biasE, encWih, idx_lds, hbuf, lane, wave, c, hout);
    __syncthreads();                                  // all reads of h_prev done
    #pragma unroll
    for (int t = 0; t < 2; t++)
      #pragma unroll
      for (int r = 0; r < 8; r++)
        hbuf[(r + mhi) * H_SZ + u0 + t * 16 + n] = f2bf(hout[t][r]);
  }

  // ---------------- Decoder: 512 steps, (h,c) carried over ----------------
  for (int s = 0; s < S_SZ; s++) {
    if (tid < TILE_M) idx_lds[tid] = E[(b0 + tid) * (S_SZ + 1) + s]; // input == target
    __syncthreads();
    lstm_wave_step(decWhh, biasD, decWih, idx_lds, hbuf, lane, wave, c, hout);
    __syncthreads();
    #pragma unroll
    for (int t = 0; t < 2; t++)
      #pragma unroll
      for (int r = 0; r < 8; r++)
        hbuf[(r + mhi) * H_SZ + u0 + t * 16 + n] = f2bf(hout[t][r]);
    __syncthreads();                                  // h_new visible for projection

    // Projection: logits[16,128] = h @ proj_W.T + proj_b  (8 waves, one 16-col tile each)
    if (wave < 8) {
      const int v0 = wave * 16;
      v8f pacc;
      const float pb = projB[v0 + n];
      #pragma unroll
      for (int r = 0; r < 8; r++) pacc[r] = pb;
      #pragma unroll 4
      for (int kk = 0; kk < 16; kk++) {
        v16bf a = load_a_frag(hbuf, lane, kk);
        v16bf b = *(const v16bf*)(projW + (size_t)(v0 + n) * H_SZ + kk * 32 + kb);
        pacc = __builtin_amdgcn_wmma_f32_16x16x32_bf16(
            false, a, false, b, (short)0, pacc, false, false);
      }
      #pragma unroll
      for (int r = 0; r < 8; r++)
        logit_lds[(r + mhi) * V_SZ + v0 + n] = pacc[r];
    }
    __syncthreads();

    // log-softmax + NLL: wave w owns batch row m=w; 128 logits over 32 lanes
    {
      const int m = wave;
      float x[4], mx = -3.4e38f;
      #pragma unroll
      for (int jj = 0; jj < 4; jj++) {
        x[jj] = logit_lds[m * V_SZ + lane + 32 * jj];
        mx = fmaxf(mx, x[jj]);
      }
      #pragma unroll
      for (int off = 16; off > 0; off >>= 1) mx = fmaxf(mx, __shfl_down(mx, off, 32));
      mx = __shfl(mx, 0, 32);
      float se = 0.0f;
      #pragma unroll
      for (int jj = 0; jj < 4; jj++) se += __expf(x[jj] - mx);
      #pragma unroll
      for (int off = 16; off > 0; off >>= 1) se += __shfl_down(se, off, 32);
      if (lane == 0) {
        const int tgt = idx_lds[m];
        const float lse = mx + __logf(se);
        const float nll = lse - logit_lds[m * V_SZ + tgt];
        const bool valid = (tgt != 0);                // PAD == 0, ignore_index
        rownll[m] = valid ? nll : 0.0f;
        rowcnt[m] = valid ? 1.0f : 0.0f;
      }
    }
    __syncthreads();
    if (tid == 0) {
      float sn = 0.0f, sc = 0.0f;
      #pragma unroll
      for (int m = 0; m < TILE_M; m++) { sn += rownll[m]; sc += rowcnt[m]; }
      partial[(size_t)(tile * S_SZ + s) * 2 + 0] = sn;
      partial[(size_t)(tile * S_SZ + s) * 2 + 1] = sc;
    }
  }
}

__global__ void cvt_bf16_kernel(const float* __restrict__ src,
                                unsigned short* __restrict__ dst, int nelem) {
  int i = blockIdx.x * blockDim.x + threadIdx.x;
  if (i < nelem) dst[i] = f2bf(src[i]);
}

__global__ void add_bias_kernel(const float* __restrict__ a, const float* __restrict__ b,
                                float* __restrict__ o, int nelem) {
  int i = blockIdx.x * blockDim.x + threadIdx.x;
  if (i < nelem) o[i] = a[i] + b[i];
}

// loss = sum_s ( cnt_s > 0 ? sum_nll_s / max(cnt_s, 1) : 0 ), batch-global per step
__global__ void __launch_bounds__(S_SZ, 1)
finalize_kernel(const float* __restrict__ partial, float* __restrict__ out) {
  __shared__ float acc[S_SZ];
  const int s = threadIdx.x;
  float sn = 0.0f, sc = 0.0f;
  #pragma unroll
  for (int t = 0; t < N_TILES; t++) {
    sn += partial[(size_t)(t * S_SZ + s) * 2 + 0];
    sc += partial[(size_t)(t * S_SZ + s) * 2 + 1];
  }
  acc[s] = (sc > 0.0f) ? (sn / fmaxf(sc, 1.0f)) : 0.0f;
  __syncthreads();
  for (int off = S_SZ / 2; off > 0; off >>= 1) {
    if (s < off) acc[s] += acc[s + off];
    __syncthreads();
  }
  if (s == 0) out[0] = acc[0];
}

extern "C" void kernel_launch(void* const* d_in, const int* in_sizes, int n_in,
                              void* d_out, int out_size, void* d_ws, size_t ws_size,
                              hipStream_t stream) {
  const int*   C_idx  = (const int*)  d_in[0];
  const int*   E      = (const int*)  d_in[1];
  const float* encWih = (const float*)d_in[2];
  const float* encWhh = (const float*)d_in[3];
  const float* encBih = (const float*)d_in[4];
  const float* encBhh = (const float*)d_in[5];
  const float* decWih = (const float*)d_in[6];
  const float* decWhh = (const float*)d_in[7];
  const float* decBih = (const float*)d_in[8];
  const float* decBhh = (const float*)d_in[9];
  const float* projW  = (const float*)d_in[10];
  const float* projB  = (const float*)d_in[11];

  // Workspace layout (~4.4 MB): bf16 weights, fused biases, per-(tile,step) loss partials
  unsigned short* encWhh_bf = (unsigned short*)d_ws;
  unsigned short* decWhh_bf = encWhh_bf + (size_t)G_SZ * H_SZ;
  unsigned short* projW_bf  = decWhh_bf + (size_t)G_SZ * H_SZ;
  float* biasE   = (float*)(projW_bf + (size_t)V_SZ * H_SZ);
  float* biasD   = biasE + G_SZ;
  float* partial = biasD + G_SZ;

  const int nWhh = G_SZ * H_SZ;
  const int nP   = V_SZ * H_SZ;
  cvt_bf16_kernel<<<(nWhh + 255) / 256, 256, 0, stream>>>(encWhh, encWhh_bf, nWhh);
  cvt_bf16_kernel<<<(nWhh + 255) / 256, 256, 0, stream>>>(decWhh, decWhh_bf, nWhh);
  cvt_bf16_kernel<<<(nP + 255) / 256, 256, 0, stream>>>(projW, projW_bf, nP);
  add_bias_kernel<<<(G_SZ + 255) / 256, 256, 0, stream>>>(encBih, encBhh, biasE, G_SZ);
  add_bias_kernel<<<(G_SZ + 255) / 256, 256, 0, stream>>>(decBih, decBhh, biasD, G_SZ);

  seq2seq_loss_kernel<<<N_TILES, N_THREADS, 0, stream>>>(
      C_idx, E, encWih, decWih, encWhh_bf, decWhh_bf, projW_bf,
      biasE, biasD, projB, partial);

  finalize_kernel<<<1, S_SZ, 0, stream>>>(partial, (float*)d_out);
}
